// GroupedQueryAttention_24120536334626
// MI455X (gfx1250) — compile-verified, matrix operand order checked by reading, not run on hardware
//
#include <hip/hip_runtime.h>
#include <hip/hip_bf16.h>

// ---------------------------------------------------------------------------
// GQA attention block for MI455X (gfx1250, wave32, WMMA).
//   x(1,2048,2048) -> Q/K/V proj -> RoPE -> causal flash attention -> Wo proj
// All GEMMs run on v_wmma_f32_16x16x32_f16 (f16 in, f32 accumulate).
// GEMMs stage 128x32 A/B panels through LDS (double-buffered, 1 barrier per
// k-step) for block-level reuse: ~64 FLOP/B from L2 instead of ~21, since
// multi-PFLOPS WMMA peak cannot be fed from L2 at per-wave reuse levels.
// Attention uses 64-key tiles so the ds_bpermute butterfly reductions of the
// online softmax are amortized over 2x the WMMA work.
// ---------------------------------------------------------------------------

typedef __attribute__((ext_vector_type(16))) _Float16 v16h;
typedef __attribute__((ext_vector_type(8)))  _Float16 v8h;
typedef __attribute__((ext_vector_type(8)))  float    v8f;

#define T_SEQ 2048
#define E_DIM 2048
#define NH    32
#define NKV   8
#define HD    64
#define KVW   (NKV * HD)   // 512
#define LDSPAD 40          // padded LDS row (halfs): 80B -> rows spread banks

union FragU { v16h v; v8h h[2]; };

// 16x32 f16 operand fragment (A-layout; also B via Bt rows) from row-major.
// Per CDNA5 ISA 7.12.2:
//   lanes 0-15 : row = row0+lane,    K = {0..7, 16..23}
//   lanes 16-31: row = row0+lane-16, K = {8..15, 24..31}
__device__ __forceinline__ v16h load_frag_g(const _Float16* __restrict__ base,
                                            int ld, int row0, int k0, int lane) {
  const int r    = row0 + (lane & 15);
  const int koff = (lane >> 4) << 3;
  const _Float16* p = base + (size_t)r * ld + (k0 + koff);
  FragU f;
  f.h[0] = *(const v8h*)(p);
  f.h[1] = *(const v8h*)(p + 16);
  return f.v;
}

__device__ __forceinline__ v16h load_frag_lds(const _Float16* base, int ld,
                                              int row0, int k0, int lane) {
  const int r    = row0 + (lane & 15);
  const int koff = (lane >> 4) << 3;
  const _Float16* p = base + r * ld + (k0 + koff);
  FragU f;
  f.h[0] = *(const v8h*)(p);
  f.h[1] = *(const v8h*)(p + 16);
  return f.v;
}

__device__ __forceinline__ v8f wmma_f16(v16h a, v16h b, v8f c) {
  return __builtin_amdgcn_wmma_f32_16x16x32_f16(false, a, false, b,
                                                (short)0, c, false, false);
}

// ---------------------------------------------------------------------------
// Elementwise f32 -> f16 cast
// ---------------------------------------------------------------------------
__global__ void cast_f16_k(const float* __restrict__ X,
                           _Float16* __restrict__ Xh, int n) {
  int i = blockIdx.x * blockDim.x + threadIdx.x;
  if (i < n) Xh[i] = (_Float16)X[i];
}

// ---------------------------------------------------------------------------
// W (K x N, f32, row-major) -> Wt (N x K, f16, row-major), LDS-tiled.
// ---------------------------------------------------------------------------
__global__ __launch_bounds__(256) void transpose_cast_k(
    const float* __restrict__ W, _Float16* __restrict__ Wt, int K, int N) {
  __shared__ float tile[32][33];
  const int bn = blockIdx.x * 32;   // along N
  const int bk = blockIdx.y * 32;   // along K
  const int tx = threadIdx.x & 31;
  const int ty = threadIdx.x >> 5;  // 0..7
#pragma unroll
  for (int i = 0; i < 32; i += 8)
    tile[ty + i][tx] = W[(size_t)(bk + ty + i) * N + bn + tx];
  __syncthreads();
#pragma unroll
  for (int i = 0; i < 32; i += 8)
    Wt[(size_t)(bn + ty + i) * K + bk + tx] = (_Float16)tile[tx][ty + i];
}

// ---------------------------------------------------------------------------
// WMMA GEMM: C(MxN) = A(MxK, f16 rowmajor) * B, with B given as Bt (NxK, f16).
// Block = 256 threads = 8 waves, block tile 128x128, wave tile 32x64.
// LDS double-buffered 128x32 panels, one __syncthreads per k-step; the next
// panel's global loads are issued before the barrier to hide latency.
// EPI: 0 = store f32 (final projection)
//      1 = RoPE epilogue, store f16 rowmajor (Q and K projections)
//      2 = store f16 transposed (N x M) (V projection -> Vt for P@V WMMA)
// ---------------------------------------------------------------------------
template<int EPI>
__global__ __launch_bounds__(256) void gemm_wmma_k(
    const _Float16* __restrict__ A, const _Float16* __restrict__ Bt,
    float* __restrict__ Cf, _Float16* __restrict__ Ch,
    int M, int N, int K,
    const float* __restrict__ fcos, const float* __restrict__ fsin) {
  __shared__ __align__(16) _Float16 ldsA[2][128 * LDSPAD];
  __shared__ __align__(16) _Float16 ldsB[2][128 * LDSPAD];

  const int lane = threadIdx.x & 31;
  const int wid  = threadIdx.x >> 5;
  const int wm   = wid >> 1;          // 0..3
  const int wn   = wid & 1;           // 0..1
  const int row0 = blockIdx.x * 128 + wm * 32;
  const int col0 = blockIdx.y * 128 + wn * 64;

  // Panel loader role of this thread: row r (0..127), half-chunk ko (0 or 16).
  const int pr = threadIdx.x >> 1;
  const int ko = (threadIdx.x & 1) << 4;
  const _Float16* gA = A  + (size_t)(blockIdx.x * 128 + pr) * K + ko;
  const _Float16* gB = Bt + (size_t)(blockIdx.y * 128 + pr) * K + ko;

  v8f c[2][4];
#pragma unroll
  for (int i = 0; i < 2; ++i)
#pragma unroll
    for (int j = 0; j < 4; ++j) { v8f z = {}; c[i][j] = z; }

  const int nk = K / 32;
  v8h pa0 = *(const v8h*)(gA);
  v8h pa1 = *(const v8h*)(gA + 8);
  v8h pb0 = *(const v8h*)(gB);
  v8h pb1 = *(const v8h*)(gB + 8);
  {
    _Float16* sa = &ldsA[0][pr * LDSPAD + ko];
    _Float16* sb = &ldsB[0][pr * LDSPAD + ko];
    *(v8h*)(sa) = pa0; *(v8h*)(sa + 8) = pa1;
    *(v8h*)(sb) = pb0; *(v8h*)(sb + 8) = pb1;
  }

  for (int kt = 0; kt < nk; ++kt) {
    const int buf = kt & 1;
    if (kt + 1 < nk) {            // issue next panel's global loads pre-barrier
      const _Float16* na = gA + (kt + 1) * 32;
      const _Float16* nb = gB + (kt + 1) * 32;
      __builtin_prefetch(na + 64, 0, 1);   // hint panel kt+3 (global_prefetch_b8)
      pa0 = *(const v8h*)(na);
      pa1 = *(const v8h*)(na + 8);
      pb0 = *(const v8h*)(nb);
      pb1 = *(const v8h*)(nb + 8);
    }
    __syncthreads();              // panel kt resident in LDS for all waves

    v16h a0 = load_frag_lds(ldsA[buf], LDSPAD, wm * 32,      0, lane);
    v16h a1 = load_frag_lds(ldsA[buf], LDSPAD, wm * 32 + 16, 0, lane);
    v16h b[4];
#pragma unroll
    for (int j = 0; j < 4; ++j)
      b[j] = load_frag_lds(ldsB[buf], LDSPAD, wn * 64 + j * 16, 0, lane);
#pragma unroll
    for (int i = 0; i < 2; ++i)
#pragma unroll
      for (int j = 0; j < 4; ++j)
        c[i][j] = wmma_f16(i ? a1 : a0, b[j], c[i][j]);

    if (kt + 1 < nk) {            // stage next panel into the other buffer
      _Float16* sa = &ldsA[buf ^ 1][pr * LDSPAD + ko];
      _Float16* sb = &ldsB[buf ^ 1][pr * LDSPAD + ko];
      *(v8h*)(sa) = pa0; *(v8h*)(sa + 8) = pa1;
      *(v8h*)(sb) = pb0; *(v8h*)(sb + 8) = pb1;
    }
  }

  // Epilogue. C layout (f32 16x16): lane 0-15 -> col=lane, row=e; lanes
  // 16-31 -> col=lane-16, row=e+8 (e = vector element index).
#pragma unroll
  for (int i = 0; i < 2; ++i) {
#pragma unroll
    for (int j = 0; j < 4; ++j) {
      const int col = col0 + j * 16 + (lane & 15);
#pragma unroll
      for (int e = 0; e < 8; ++e) {
        const int row = row0 + i * 16 + e + ((lane >> 4) << 3);
        float v = c[i][j][e];
        if (EPI == 0) {
          Cf[(size_t)row * N + col] = v;
        } else if (EPI == 1) {
          // RoPE on (even,odd) interleaved pairs; partner value sits in the
          // adjacent lane of the same 16-lane half -> wave32 shfl_xor(1).
          float partner = __shfl_xor(v, 1, 32);
          const int d  = col & (HD - 1);
          const int fi = d >> 1;
          const float cs = fcos[row * (HD / 2) + fi];
          const float sn = fsin[row * (HD / 2) + fi];
          float out = ((col & 1) == 0) ? (v * cs - partner * sn)
                                       : (partner * sn + v * cs);
          Ch[(size_t)row * N + col] = (_Float16)out;
        } else { // EPI == 2 : transposed store, Ch is Vt (N x M)
          Ch[(size_t)col * M + row] = (_Float16)v;
        }
      }
    }
  }
}

// ---------------------------------------------------------------------------
// Flash attention, causal, GQA (4 Q-heads per KV head).
// 1 wave per (head, 16-query tile). Block = 128 threads = 4 waves.
// 64-key tiles: one butterfly-reduction pass (ds_bpermute) and one LDS P
// round-trip per 16 WMMAs. attn scale * log2(e) folded into the resident Q
// fragments; softmax runs in the exp2 domain. Exactly one causally-partial
// tile per query tile (qrow0 mod 64 in {0,16,32,48}); all earlier tiles are
// mask-free.
//   Qh : T x 2048 f16 (RoPE'd), Kh : T x 512 f16 (RoPE'd)
//   Vt : 512 x T f16 (d-major -> contiguous B fragments for P@V)
//   Yh : T x 2048 f16 (attention output, input to Wo GEMM)
// ---------------------------------------------------------------------------
__global__ __launch_bounds__(128) void attn_k(
    const _Float16* __restrict__ Qh, const _Float16* __restrict__ Kh,
    const _Float16* __restrict__ Vt, _Float16* __restrict__ Yh) {
  __shared__ __align__(16) _Float16 lds_p[4][16 * 64];
  const int lane = threadIdx.x & 31;
  const int wid  = threadIdx.x >> 5;
  const int qt   = blockIdx.x * 4 + wid;   // 16-row query tile index
  const int h    = blockIdx.y;             // Q head
  const int kv   = h >> 2;                 // KV head (REPEATS = 4)
  const int qrow0 = qt * 16;

  // Q fragments (16 x 64), scaled by 64^-0.5 * log2(e), resident all loop.
  FragU uq0, uq1;
  uq0.v = load_frag_g(Qh + h * HD, E_DIM, qrow0, 0,  lane);
  uq1.v = load_frag_g(Qh + h * HD, E_DIM, qrow0, 32, lane);
  const _Float16 qsc = (_Float16)0.18033688f;   // 0.125 * log2(e)
#pragma unroll
  for (int i = 0; i < 16; ++i) { uq0.v[i] *= qsc; uq1.v[i] *= qsc; }
  const v16h qa0 = uq0.v, qa1 = uq1.v;

  v8f o[4];
#pragma unroll
  for (int dt = 0; dt < 4; ++dt) { v8f z = {}; o[dt] = z; }
  float mrow[8], lrow[8];
#pragma unroll
  for (int e = 0; e < 8; ++e) { mrow[e] = -1e30f; lrow[e] = 0.0f; }

  auto process_tile = [&](int kb, bool mask) {
    // S(16x64) = Q(16x64) * K^T : Kh rows are exactly the B^T rows we need.
    v8f s[4];
#pragma unroll
    for (int nn = 0; nn < 4; ++nn) {
      v16h b0 = load_frag_g(Kh + kv * HD, KVW, kb + nn * 16, 0,  lane);
      v16h b1 = load_frag_g(Kh + kv * HD, KVW, kb + nn * 16, 32, lane);
      v8f z = {};
      z = wmma_f16(qa0, b0, z);
      s[nn] = wmma_f16(qa1, b1, z);
    }
    if (mask) {   // causal mask, only on the single partial tile
#pragma unroll
      for (int nn = 0; nn < 4; ++nn)
#pragma unroll
        for (int e = 0; e < 8; ++e) {
          const int key = kb + nn * 16 + (lane & 15);
          const int qr  = qrow0 + e + ((lane >> 4) << 3);
          s[nn][e] = (key <= qr) ? s[nn][e] : -1e30f;
        }
    }
    // Online softmax (exp2 domain), per row e across the 16-lane half.
#pragma unroll
    for (int e = 0; e < 8; ++e) {
      float rm = fmaxf(fmaxf(s[0][e], s[1][e]), fmaxf(s[2][e], s[3][e]));
#pragma unroll
      for (int off = 1; off < 16; off <<= 1)
        rm = fmaxf(rm, __shfl_xor(rm, off, 32));
      const float mnew  = fmaxf(mrow[e], rm);
      const float alpha = exp2f(mrow[e] - mnew);
      mrow[e] = mnew;
      float rs = 0.0f;
#pragma unroll
      for (int nn = 0; nn < 4; ++nn) {
        const float p = exp2f(s[nn][e] - mnew);
        s[nn][e] = p;
        rs += p;
      }
#pragma unroll
      for (int off = 1; off < 16; off <<= 1)
        rs += __shfl_xor(rs, off, 32);
      lrow[e] = lrow[e] * alpha + rs;
#pragma unroll
      for (int dt = 0; dt < 4; ++dt) o[dt][e] *= alpha;
    }
    // Re-layout P (C layout f32) -> A fragments (f16) through LDS.
#pragma unroll
    for (int nn = 0; nn < 4; ++nn)
#pragma unroll
      for (int e = 0; e < 8; ++e) {
        const int r = e + ((lane >> 4) << 3);
        const int cidx = nn * 16 + (lane & 15);
        lds_p[wid][r * 64 + cidx] = (_Float16)s[nn][e];
      }
    asm volatile("s_wait_dscnt 0" ::: "memory");  // same-wave LDS RAW fence
    const v16h pf0 = load_frag_lds(lds_p[wid], 64, 0, 0,  lane);
    const v16h pf1 = load_frag_lds(lds_p[wid], 64, 0, 32, lane);
    // O(16x64) += P(16x64) * V(64x64); Vt rows are the B^T rows, contiguous.
#pragma unroll
    for (int dt = 0; dt < 4; ++dt) {
      v16h bv0 = load_frag_g(Vt, T_SEQ, kv * HD + dt * 16, kb,      lane);
      v16h bv1 = load_frag_g(Vt, T_SEQ, kv * HD + dt * 16, kb + 32, lane);
      o[dt] = wmma_f16(pf1, bv1, wmma_f16(pf0, bv0, o[dt]));
    }
  };

  const int nfull = qrow0 >> 6;            // fully-unmasked 64-key tiles
  for (int kt = 0; kt < nfull; ++kt) process_tile(kt * 64, false);
  process_tile(nfull * 64, true);          // exactly one causal-partial tile

  // Normalize and store Y (f16, row-major T x 2048).
#pragma unroll
  for (int e = 0; e < 8; ++e) {
    const float inv = 1.0f / lrow[e];
    const int row = qrow0 + e + ((lane >> 4) << 3);
#pragma unroll
    for (int dt = 0; dt < 4; ++dt) {
      const int col = h * HD + dt * 16 + (lane & 15);
      Yh[(size_t)row * E_DIM + col] = (_Float16)(o[dt][e] * inv);
    }
  }
}

// ---------------------------------------------------------------------------
// Launcher. Workspace layout (f16 unless noted), 48 MiB total:
//   xh 8M | WqT 8M | WkT 2M | WvT 2M | WoT 8M | Qh 8M | Kh 2M | Vt 2M | Yh 8M
// ---------------------------------------------------------------------------
extern "C" void kernel_launch(void* const* d_in, const int* in_sizes, int n_in,
                              void* d_out, int out_size, void* d_ws, size_t ws_size,
                              hipStream_t stream) {
  const float* x    = (const float*)d_in[0];
  const float* fcos = (const float*)d_in[1];
  const float* fsin = (const float*)d_in[2];
  const float* Wq   = (const float*)d_in[3];
  const float* Wk   = (const float*)d_in[4];
  const float* Wv   = (const float*)d_in[5];
  const float* Wo   = (const float*)d_in[6];
  float* out = (float*)d_out;

  char* w = (char*)d_ws;
  const size_t MB = 1u << 20;
  _Float16* xh  = (_Float16*)w;            w += 8 * MB;
  _Float16* WqT = (_Float16*)w;            w += 8 * MB;
  _Float16* WkT = (_Float16*)w;            w += 2 * MB;
  _Float16* WvT = (_Float16*)w;            w += 2 * MB;
  _Float16* WoT = (_Float16*)w;            w += 8 * MB;
  _Float16* Qh  = (_Float16*)w;            w += 8 * MB;
  _Float16* Kh  = (_Float16*)w;            w += 2 * MB;
  _Float16* Vt  = (_Float16*)w;            w += 2 * MB;
  _Float16* Yh  = (_Float16*)w;            w += 8 * MB;

  const int nx = T_SEQ * E_DIM;
  cast_f16_k<<<(nx + 255) / 256, 256, 0, stream>>>(x, xh, nx);

  transpose_cast_k<<<dim3(E_DIM / 32, E_DIM / 32), 256, 0, stream>>>(Wq, WqT, E_DIM, E_DIM);
  transpose_cast_k<<<dim3(KVW   / 32, E_DIM / 32), 256, 0, stream>>>(Wk, WkT, E_DIM, KVW);
  transpose_cast_k<<<dim3(KVW   / 32, E_DIM / 32), 256, 0, stream>>>(Wv, WvT, E_DIM, KVW);
  transpose_cast_k<<<dim3(E_DIM / 32, E_DIM / 32), 256, 0, stream>>>(Wo, WoT, E_DIM, E_DIM);

  // Q/K projections with fused RoPE; V projection stored transposed.
  gemm_wmma_k<1><<<dim3(T_SEQ / 128, E_DIM / 128), 256, 0, stream>>>(
      xh, WqT, nullptr, Qh, T_SEQ, E_DIM, E_DIM, fcos, fsin);
  gemm_wmma_k<1><<<dim3(T_SEQ / 128, KVW / 128), 256, 0, stream>>>(
      xh, WkT, nullptr, Kh, T_SEQ, KVW, E_DIM, fcos, fsin);
  gemm_wmma_k<2><<<dim3(T_SEQ / 128, KVW / 128), 256, 0, stream>>>(
      xh, WvT, nullptr, Vt, T_SEQ, KVW, E_DIM, nullptr, nullptr);

  // Causal GQA flash attention: grid = (qtiles/4, heads).
  attn_k<<<dim3(T_SEQ / 16 / 4, NH), 128, 0, stream>>>(Qh, Kh, Vt, Yh);

  // Output projection -> f32.
  gemm_wmma_k<0><<<dim3(T_SEQ / 128, E_DIM / 128), 256, 0, stream>>>(
      Yh, WoT, out, nullptr, T_SEQ, E_DIM, E_DIM, nullptr, nullptr);
}
